// PALMAttention_56942676410743
// MI455X (gfx1250) — matrix operand order repaired and checked by reading, not compiled, not run on hardware
//
#include <hip/hip_runtime.h>
#include <hip/hip_bf16.h>

typedef __attribute__((ext_vector_type(16))) _Float16 v16h;
typedef __attribute__((ext_vector_type(8)))  _Float16 v8h;
typedef __attribute__((ext_vector_type(4)))  _Float16 v4h;
typedef __attribute__((ext_vector_type(8)))  float    v8f;

union F16x16 { v16h v; v8h h[2]; };

#if defined(__has_builtin)
#if __has_builtin(__builtin_amdgcn_global_load_async_to_lds_b128)
#define HAVE_ASYNC_LDS 1
#endif
#endif
#ifndef HAVE_ASYNC_LDS
#define HAVE_ASYNC_LDS 0
#endif

// ROCm clang-22 declares the async-LDS builtins with generic `int4*` params
// (probe-confirmed via diagnostic: "__attribute__((__vector_size__(4 *
// sizeof(int)))) int __device__ *").
typedef int v4i_vs __attribute__((vector_size(16)));

// 16-byte global -> LDS copy. Async (ASYNCcnt-tracked, no VGPR round trip)
// when the gfx1250 builtin is available, else plain load+ds_store.
__device__ __forceinline__ void cp_async16(const _Float16* g, _Float16* l) {
#if HAVE_ASYNC_LDS
  __builtin_amdgcn_global_load_async_to_lds_b128(
      (v4i_vs*)(void*)const_cast<_Float16*>(g), (v4i_vs*)(void*)l, 0, 0);
#else
  *(v8h*)l = *(const v8h*)g;
#endif
}

#if HAVE_ASYNC_LDS
// 12 async ops per stage per thread: wait "<=12" keeps next stage in flight.
#define ASYNC_WAIT_SOME() asm volatile("s_wait_asynccnt 0xc" ::: "memory")
#define ASYNC_WAIT_ALL()  asm volatile("s_wait_asynccnt 0x0" ::: "memory")
#else
#define ASYNC_WAIT_SOME() asm volatile("s_wait_dscnt 0x0" ::: "memory")
#define ASYNC_WAIT_ALL()  asm volatile("s_wait_dscnt 0x0" ::: "memory")
#endif

// A-fragment (16x32 f16, M x K): lane l holds row (l&15); lanes 0-15 hold
// K=0..7 & 16..23, lanes 16-31 hold K=8..15 & 24..31. p is pre-offset to
// row*stride + (lane>=16 ? 8 : 0).
__device__ __forceinline__ v16h load_fragA(const _Float16* p) {
  F16x16 f;
  f.h[0] = *(const v8h*)(p);
  f.h[1] = *(const v8h*)(p + 16);
  return f.v;
}
// B-fragment (32x16 f16, K x N): lane l holds column (l&15); lanes 0-15 hold
// K=0..7 & 16..23, lanes 16-31 hold K=8..15 & 24..31 (same as A).
__device__ __forceinline__ v16h load_fragB(const _Float16* p) {
  F16x16 f;
  f.h[0] = *(const v8h*)(p);
  f.h[1] = *(const v8h*)(p + 16);
  return f.v;
}
__device__ __forceinline__ v8f wmma16(v16h a, v16h b, v8f c) {
  return __builtin_amdgcn_wmma_f32_16x16x32_f16(false, a, false, b, (short)0, c,
                                                false, false);
}

// ---------------------------------------------------------------- f32 -> f16
__global__ __launch_bounds__(256) void k_cvt4(const float* __restrict__ s,
                                              _Float16* __restrict__ d, int n4) {
  int i = blockIdx.x * 256 + threadIdx.x;
  if (i < n4) {
    float4 v = ((const float4*)s)[i];
    v4h o;
    o[0] = (_Float16)v.x; o[1] = (_Float16)v.y;
    o[2] = (_Float16)v.z; o[3] = (_Float16)v.w;
    ((v4h*)d)[i] = o;
  }
}

// --------------------------------------------------------------------- GEMM
// out[m,n] = sum_k A[m,k]*Bw[n,k] + bias[n]  (A: M x K, Bw: N x K row-major)
// Block tile 128x64, K-step 64, double-buffered async LDS staging.
// mode 0: out16 at q/k head layout [b,h,s,d]
// mode 1: out16 at transposed-V layout [b,h,d,s]
// mode 2: outf = val + resid (f32) (final projection, pre-layernorm)
__device__ __forceinline__ void stage_tiles(const _Float16* __restrict__ A,
                                            const _Float16* __restrict__ Bw,
                                            _Float16* As, _Float16* Bs, int m0,
                                            int n0, int K, int kb, int t) {
  // A tile: 128 rows x 64 halves. Thread t copies row t (8 x 16B).
  const _Float16* ga = A + (size_t)(m0 + t) * K + kb;
  _Float16* la = As + t * 64;
#pragma unroll
  for (int c = 0; c < 8; ++c) cp_async16(ga + c * 8, la + c * 8);
  // B tile: 64 rows x 64 halves. Thread t copies half-row (4 x 16B).
  const int brow = t >> 1, bcol = (t & 1) * 32;
  const _Float16* gb = Bw + (size_t)(n0 + brow) * K + kb + bcol;
  _Float16* lb = Bs + brow * 64 + bcol;
#pragma unroll
  for (int c = 0; c < 4; ++c) cp_async16(gb + c * 8, lb + c * 8);
}

__global__ __launch_bounds__(128) void k_gemm(
    const _Float16* __restrict__ A, const _Float16* __restrict__ Bw,
    const float* __restrict__ bias, const float* __restrict__ resid,
    _Float16* __restrict__ out16, float* __restrict__ outf, int K, int mode) {
  __shared__ __align__(16) _Float16 As[2][128 * 64];  // 32 KB
  __shared__ __align__(16) _Float16 Bs[2][64 * 64];   // 16 KB
  const int m0 = blockIdx.x * 128, n0 = blockIdx.y * 64;
  const int t = threadIdx.x, wave = t >> 5, lane = t & 31;
  const int hl = lane >> 4, ln = lane & 15;
  v8f acc[2][4] = {};

  stage_tiles(A, Bw, &As[0][0], &Bs[0][0], m0, n0, K, 0, t);
  const int nk = K >> 6;
  for (int it = 0; it < nk; ++it) {
    const int bf = it & 1;
    if (it + 1 < nk) {
      stage_tiles(A, Bw, &As[bf ^ 1][0], &Bs[bf ^ 1][0], m0, n0, K,
                  (it + 1) * 64, t);
      ASYNC_WAIT_SOME();
    } else {
      ASYNC_WAIT_ALL();
    }
    __syncthreads();
    const _Float16* as = &As[bf][0];
    const _Float16* bs = &Bs[bf][0];
#pragma unroll
    for (int ks = 0; ks < 2; ++ks) {
      v16h af0 = load_fragA(as + (wave * 32 + ln) * 64 + ks * 32 + hl * 8);
      v16h af1 = load_fragA(as + (wave * 32 + 16 + ln) * 64 + ks * 32 + hl * 8);
      v16h bfr[4];
#pragma unroll
      for (int j = 0; j < 4; ++j)
        bfr[j] = load_fragB(bs + (j * 16 + ln) * 64 + ks * 32 + hl * 8);
#pragma unroll
      for (int j = 0; j < 4; ++j) {
        acc[0][j] = wmma16(af0, bfr[j], acc[0][j]);
        acc[1][j] = wmma16(af1, bfr[j], acc[1][j]);
      }
    }
    __syncthreads();
  }
  // C/D layout: VGPR r, lanes 0-15 -> (M=r, N=lane); lanes 16-31 -> (M=r+8).
#pragma unroll
  for (int rr = 0; rr < 2; ++rr) {
#pragma unroll
    for (int j = 0; j < 4; ++j) {
#pragma unroll
      for (int r = 0; r < 8; ++r) {
        int n = n0 + j * 16 + ln;
        int m = m0 + wave * 32 + rr * 16 + r + hl * 8;
        float val = acc[rr][j][r] + bias[n];
        if (mode == 2) {
          outf[(size_t)m * 1024 + n] = val + resid[(size_t)m * 1024 + n];
        } else {
          int b = m >> 11, s = m & 2047;
          int h = n >> 6,  d = n & 63;
          size_t idx = (mode == 0)
                           ? ((size_t)((b * 16 + h) * 2048 + s) * 64 + d)
                           : ((size_t)((b * 16 + h) * 64 + d) * 2048 + s);
          out16[idx] = (_Float16)val;
        }
      }
    }
  }
}

// ---------------------------------------------------- flash attention (WMMA)
__global__ __launch_bounds__(128) void k_attn(
    const _Float16* __restrict__ q16,   // [BH, S, 64]
    const _Float16* __restrict__ k16,   // [BH, S, 64]
    const _Float16* __restrict__ vt16,  // [BH, 64, S]
    const float* __restrict__ amask,    // [B, S]
    _Float16* __restrict__ ctx16) {     // [B, S, 1024] col = h*64+d
  const int S = 2048, HD = 64, H = 1024;
  const int bh = blockIdx.y, b = bh >> 4, h = bh & 15;
  const int t = threadIdx.x, wave = t >> 5, lane = t & 31;
  const int hl = lane >> 4, ln = lane & 15;
  const int qw = blockIdx.x * 64 + wave * 16;
  __shared__ __align__(16) _Float16 plds[4][16 * 64];

  const _Float16* qbase = q16 + ((size_t)bh * S + qw) * HD;
  v16h aq0 = load_fragA(qbase + ln * HD + hl * 8);       // d = 0..31
  v16h aq1 = load_fragA(qbase + ln * HD + 32 + hl * 8);  // d = 32..63

  v8f acc[4] = {};
  float mrow[8], lrow[8];
#pragma unroll
  for (int r = 0; r < 8; r++) { mrow[r] = -3.0e30f; lrow[r] = 0.0f; }
  const float sc = 0.125f;  // 1/sqrt(64)

  for (int kb = 0; kb < S; kb += 64) {
    v8f sv[4] = {};
#pragma unroll
    for (int j = 0; j < 4; j++) {
      const _Float16* kbase =
          k16 + ((size_t)bh * S + kb + j * 16 + ln) * HD + hl * 8;
      v16h bk0 = load_fragB(kbase);
      v16h bk1 = load_fragB(kbase + 32);
      sv[j] = wmma16(aq0, bk0, sv[j]);
      sv[j] = wmma16(aq1, bk1, sv[j]);
    }
    float madd[4];
#pragma unroll
    for (int j = 0; j < 4; j++)
      madd[j] = (1.0f - amask[b * S + kb + j * 16 + ln]) * -10000.0f;
    _Float16* pw = &plds[wave][0];
#pragma unroll
    for (int r = 0; r < 8; r++) {
      float a0 = sv[0][r] * sc + madd[0];
      float a1 = sv[1][r] * sc + madd[1];
      float a2 = sv[2][r] * sc + madd[2];
      float a3 = sv[3][r] * sc + madd[3];
      float mx = fmaxf(fmaxf(a0, a1), fmaxf(a2, a3));
#pragma unroll
      for (int off = 8; off; off >>= 1) mx = fmaxf(mx, __shfl_xor(mx, off, 16));
      float mnew = fmaxf(mrow[r], mx);
      float corr = __expf(mrow[r] - mnew);
      float p0 = __expf(a0 - mnew);
      float p1 = __expf(a1 - mnew);
      float p2 = __expf(a2 - mnew);
      float p3 = __expf(a3 - mnew);
      float ps = (p0 + p1) + (p2 + p3);
#pragma unroll
      for (int off = 8; off; off >>= 1) ps += __shfl_xor(ps, off, 16);
      lrow[r] = lrow[r] * corr + ps;
      mrow[r] = mnew;
      acc[0][r] *= corr; acc[1][r] *= corr;
      acc[2][r] *= corr; acc[3][r] *= corr;
      int prow = r + hl * 8;  // transpose C-layout -> row-major p in LDS
      pw[prow * 64 + ln]      = (_Float16)p0;
      pw[prow * 64 + 16 + ln] = (_Float16)p1;
      pw[prow * 64 + 32 + ln] = (_Float16)p2;
      pw[prow * 64 + 48 + ln] = (_Float16)p3;
    }
    asm volatile("s_wait_dscnt 0x0" ::: "memory");  // per-wave LDS RAW
    v16h ap0 = load_fragA(pw + ln * 64 + hl * 8);       // keys 0..31
    v16h ap1 = load_fragA(pw + ln * 64 + 32 + hl * 8);  // keys 32..63
#pragma unroll
    for (int j = 0; j < 4; j++) {
      const _Float16* vb =
          vt16 + ((size_t)bh * HD + j * 16 + ln) * S + kb + hl * 8;
      v16h bv0 = load_fragB(vb);
      v16h bv1 = load_fragB(vb + 32);
      acc[j] = wmma16(ap0, bv0, acc[j]);
      acc[j] = wmma16(ap1, bv1, acc[j]);
    }
  }
#pragma unroll
  for (int j = 0; j < 4; j++) {
#pragma unroll
    for (int r = 0; r < 8; r++) {
      float val = acc[j][r] / lrow[r];
      int row = qw + r + hl * 8;
      int col = h * 64 + j * 16 + ln;
      ctx16[((size_t)b * S + row) * H + col] = (_Float16)val;
    }
  }
}

// ------------------------------------------------------------ layernorm row
__device__ __forceinline__ float wave_sum(float v) {
#pragma unroll
  for (int off = 16; off; off >>= 1) v += __shfl_xor(v, off, 32);
  return v;
}

__global__ __launch_bounds__(256) void k_ln(const float* __restrict__ y,
                                            const float* __restrict__ g,
                                            const float* __restrict__ be,
                                            float* __restrict__ out) {
  const int H = 1024;
  const size_t row = blockIdx.x;
  const int t = threadIdx.x, wid = t >> 5, lane = t & 31;
  __shared__ float sm[8];
  const float* yr = y + row * H;
  float v[4];
  float s = 0.0f;
#pragma unroll
  for (int i = 0; i < 4; i++) { v[i] = yr[t + i * 256]; s += v[i]; }
  s = wave_sum(s);
  if (lane == 0) sm[wid] = s;
  __syncthreads();
  float tot = 0.0f;
#pragma unroll
  for (int i = 0; i < 8; i++) tot += sm[i];
  float mean = tot * (1.0f / 1024.0f);
  __syncthreads();
  float vs = 0.0f;
#pragma unroll
  for (int i = 0; i < 4; i++) { float d = v[i] - mean; vs += d * d; }
  vs = wave_sum(vs);
  if (lane == 0) sm[wid] = vs;
  __syncthreads();
  float vtot = 0.0f;
#pragma unroll
  for (int i = 0; i < 8; i++) vtot += sm[i];
  float rstd = rsqrtf(vtot * (1.0f / 1024.0f) + 1e-12f);
#pragma unroll
  for (int i = 0; i < 4; i++) {
    int c = t + i * 256;
    float o = (v[i] - mean) * rstd * g[c] + be[c];
    out[row * H + c] = __builtin_isfinite(o) ? o : 0.0f;
  }
}

// ------------------------------------------------------------------- launch
extern "C" void kernel_launch(void* const* d_in, const int* in_sizes, int n_in,
                              void* d_out, int out_size, void* d_ws,
                              size_t ws_size, hipStream_t stream) {
  const float* x  = (const float*)d_in[0];
  const float* am = (const float*)d_in[1];
  const float* Wq = (const float*)d_in[2];
  const float* bq = (const float*)d_in[3];
  const float* Wk = (const float*)d_in[4];
  const float* bk = (const float*)d_in[5];
  const float* Wv = (const float*)d_in[6];
  const float* bv = (const float*)d_in[7];
  const float* Wd = (const float*)d_in[8];
  const float* bd = (const float*)d_in[9];
  const float* lg = (const float*)d_in[10];
  const float* lb = (const float*)d_in[11];
  float* out = (float*)d_out;

  char* ws = (char*)d_ws;
  const size_t MB = 1024 * 1024;
  _Float16* x16  = (_Float16*)(ws);             //  8 MB: x  [4096,1024] f16
  _Float16* wq16 = (_Float16*)(ws + 8 * MB);    //  2 MB
  _Float16* wk16 = (_Float16*)(ws + 10 * MB);   //  2 MB
  _Float16* wv16 = (_Float16*)(ws + 12 * MB);   //  2 MB
  _Float16* wd16 = (_Float16*)(ws + 14 * MB);   //  2 MB
  _Float16* q16  = (_Float16*)(ws + 16 * MB);   //  8 MB: [BH,S,64]
  _Float16* k16  = (_Float16*)(ws + 24 * MB);   //  8 MB: [BH,S,64]
  _Float16* vt16 = (_Float16*)(ws + 32 * MB);   //  8 MB: [BH,64,S]
  _Float16* c16  = (_Float16*)(ws + 40 * MB);   //  8 MB: ctx [4096,1024]
  float*    ybuf = (float*)(ws + 48 * MB);      // 16 MB: pre-LN f32

  k_cvt4<<<4096, 256, 0, stream>>>(x, x16, 1048576);
  k_cvt4<<<1024, 256, 0, stream>>>(Wq, wq16, 262144);
  k_cvt4<<<1024, 256, 0, stream>>>(Wk, wk16, 262144);
  k_cvt4<<<1024, 256, 0, stream>>>(Wv, wv16, 262144);
  k_cvt4<<<1024, 256, 0, stream>>>(Wd, wd16, 262144);

  dim3 gg(32, 16), gb(128);
  k_gemm<<<gg, gb, 0, stream>>>(x16, wq16, bq, nullptr, q16, nullptr, 1024, 0);
  k_gemm<<<gg, gb, 0, stream>>>(x16, wk16, bk, nullptr, k16, nullptr, 1024, 0);
  k_gemm<<<gg, gb, 0, stream>>>(x16, wv16, bv, nullptr, vt16, nullptr, 1024, 1);

  k_attn<<<dim3(32, 32), 128, 0, stream>>>(q16, k16, vt16, am, c16);

  k_gemm<<<gg, gb, 0, stream>>>(c16, wd16, bd, x, nullptr, ybuf, 1024, 2);
  k_ln<<<4096, 256, 0, stream>>>(ybuf, lg, lb, out);
}
